// STMBlock_13176959664590
// MI455X (gfx1250) — matrix operand order repaired and checked by reading, not compiled, not run on hardware
//
#include <hip/hip_runtime.h>
#include <hip/hip_bf16.h>
#include <math.h>

// ---------------------------------------------------------------------------
// CDNA5 (gfx1250) wave32 WMMA types & fragment loaders
// Layouts per CDNA5 ISA 7.12.2:
//   A (16x32 f16): lane -> row = lane&15; slot e: k = (e<8)? hi*8+e : 16+hi*8+(e-8)
//   B (32x16 f16): lane -> col = lane&15; slot e: k = (e<8)? hi*8+e : 16+hi*8+(e-8)
//   C/D (16x16 f32): lane -> col = lane&15; vgpr r -> row = r + 8*hi
// ---------------------------------------------------------------------------
typedef __attribute__((ext_vector_type(16))) _Float16 v16h;
typedef __attribute__((ext_vector_type(8)))  _Float16 v8h;
typedef __attribute__((ext_vector_type(8)))  float    v8f;

union U16H { v16h v; _Float16 e[16]; };
union U8F  { v8f  v; float    e[8];  };

__device__ __forceinline__ v16h cat8(v8h a, v8h b) {
  return __builtin_shufflevector(a, b, 0,1,2,3,4,5,6,7,8,9,10,11,12,13,14,15);
}
// A fragment: rowp points at A[row][0]; covers k in [k0, k0+32)
__device__ __forceinline__ v16h load_a16(const _Float16* rowp, int k0, int hi) {
  v8h lo = *(const v8h*)(rowp + k0 + hi * 8);
  v8h hh = *(const v8h*)(rowp + k0 + 16 + hi * 8);
  return cat8(lo, hh);
}
// B fragment: colp points at Bt[col][0] (K-contiguous); covers k in [k0, k0+32)
__device__ __forceinline__ v16h load_b16(const _Float16* colp, int k0, int hi) {
  v8h lo = *(const v8h*)(colp + k0 + hi * 8);
  v8h hh = *(const v8h*)(colp + k0 + 16 + hi * 8);
  return cat8(lo, hh);
}
__device__ __forceinline__ v8f wmma_f16(v16h a, v16h b, v8f c) {
  return __builtin_amdgcn_wmma_f32_16x16x32_f16(false, a, false, b, (short)0, c, false, false);
}
__device__ __forceinline__ float wave_sum(float v) {
  for (int m = 1; m < 32; m <<= 1) v += __shfl_xor(v, m, 32);
  return v;
}

#define STM_B   16
#define STM_N   2048
#define STM_C   384
#define STM_H   6
#define STM_DH  64
#define STM_OT  512
#define STM_FC  1536
#define INV_SQRT_DIM 0.051031036307982884f  /* 1/sqrt(384) */
#define POS_INF __int_as_float(0x7f800000)

// ---------------------------------------------------------------------------
// K1: fused LayerNorm1 + token_score + row sq-norm. 1 wave per row.
// ---------------------------------------------------------------------------
__global__ void ln_score_kernel(const float* __restrict__ x,
                                const float* __restrict__ g, const float* __restrict__ be,
                                const float* __restrict__ score_w, const float* __restrict__ score_b,
                                _Float16* __restrict__ xnH, float* __restrict__ tscore,
                                float* __restrict__ sq) {
  int lane = threadIdx.x & 31;
  long row = (long)blockIdx.x * 8 + (threadIdx.x >> 5);
  const float* xr = x + row * STM_C;
  float xv[12], s = 0.f;
  for (int j = 0; j < 12; ++j) { xv[j] = xr[lane + 32 * j]; s += xv[j]; }
  float mean = wave_sum(s) * (1.0f / STM_C);
  float v = 0.f;
  for (int j = 0; j < 12; ++j) { float d = xv[j] - mean; v += d * d; }
  float rstd = rsqrtf(wave_sum(v) * (1.0f / STM_C) + 1e-5f);
  float sv = 0.f, sc = 0.f;
  for (int j = 0; j < 12; ++j) {
    int e = lane + 32 * j;
    float xn = (xv[j] - mean) * rstd * g[e] + be[e];
    xnH[row * STM_C + e] = (_Float16)xn;
    sv += xn * xn;
    sc += xn * score_w[e];
  }
  sv = wave_sum(sv); sc = wave_sum(sc);
  if (lane == 0) { sq[row] = sv; tscore[row] = sc + score_b[0]; }
}

// LayerNorm2 for the 8192 merged-token rows.
__global__ void ln2_kernel(const float* __restrict__ x, const float* __restrict__ g,
                           const float* __restrict__ be, _Float16* __restrict__ o) {
  int lane = threadIdx.x & 31;
  long row = (long)blockIdx.x * 8 + (threadIdx.x >> 5);
  const float* xr = x + row * STM_C;
  float xv[12], s = 0.f;
  for (int j = 0; j < 12; ++j) { xv[j] = xr[lane + 32 * j]; s += xv[j]; }
  float mean = wave_sum(s) * (1.0f / STM_C);
  float v = 0.f;
  for (int j = 0; j < 12; ++j) { float d = xv[j] - mean; v += d * d; }
  float rstd = rsqrtf(wave_sum(v) * (1.0f / STM_C) + 1e-5f);
  for (int j = 0; j < 12; ++j) {
    int e = lane + 32 * j;
    o[row * STM_C + e] = (_Float16)((xv[j] - mean) * rstd * g[e] + be[e]);
  }
}

// ---------------------------------------------------------------------------
// Utility: zero fill / f32->f16 / weight transpose-convert w[K][N] -> wt[N][K]
// ---------------------------------------------------------------------------
__global__ void zero_kernel(float* p, long n) {
  long t = (long)blockIdx.x * 256 + threadIdx.x;
  if (t < n) p[t] = 0.f;
}
__global__ void f2h_kernel(const float* a, _Float16* o, long n) {
  long t = (long)blockIdx.x * 256 + threadIdx.x;
  if (t < n) o[t] = (_Float16)a[t];
}
__global__ void wtr_kernel(const float* __restrict__ w, _Float16* __restrict__ wt, int K, int N) {
  long t = (long)blockIdx.x * 256 + threadIdx.x;
  if (t >= (long)K * N) return;
  int k = (int)(t / N), n = (int)(t % N);
  wt[(size_t)n * K + k] = (_Float16)w[t];
}

// ---------------------------------------------------------------------------
// Generic register-blocked WMMA GEMM: 32x64 output per wave (2 M-tiles x 4
// N-tiles -> each A frag reused 4x, each B frag 2x: ~22 flop/byte of frag
// traffic vs 8 for a 16x16 tile). C[M,N] = A[M,K] @ Bt[N,K]^T.
// epi: 0 = store f16; 1 = outF = acc + bias + res; 2 = outH = gelu(acc + bias)
// ---------------------------------------------------------------------------
__global__ void gemm_wmma_kernel(const _Float16* __restrict__ A, const _Float16* __restrict__ Bt,
                                 int M, int N, int K,
                                 float* __restrict__ outF, _Float16* __restrict__ outH,
                                 const float* __restrict__ bias, const float* __restrict__ res,
                                 int epi) {
  int lane = threadIdx.x & 31, lo = lane & 15, hi = lane >> 4;
  int n0 = blockIdx.x * 64, m0 = blockIdx.y * 32;
  const _Float16* ar0 = A + (size_t)(m0 + lo) * K;
  const _Float16* ar1 = ar0 + (size_t)16 * K;
  const _Float16* bc0 = Bt + (size_t)(n0 + lo) * K;
  const _Float16* bc1 = bc0 + (size_t)16 * K;
  const _Float16* bc2 = bc0 + (size_t)32 * K;
  const _Float16* bc3 = bc0 + (size_t)48 * K;
  v8f acc[2][4] = {};
  for (int k0 = 0; k0 < K; k0 += 32) {
    __builtin_prefetch(ar0 + k0 + 128, 0, 1);   // global_prefetch_b8
    v16h a0 = load_a16(ar0, k0, hi);
    v16h a1 = load_a16(ar1, k0, hi);
    v16h b0 = load_b16(bc0, k0, hi);
    v16h b1 = load_b16(bc1, k0, hi);
    v16h b2 = load_b16(bc2, k0, hi);
    v16h b3 = load_b16(bc3, k0, hi);
    acc[0][0] = wmma_f16(a0, b0, acc[0][0]);
    acc[1][0] = wmma_f16(a1, b0, acc[1][0]);
    acc[0][1] = wmma_f16(a0, b1, acc[0][1]);
    acc[1][1] = wmma_f16(a1, b1, acc[1][1]);
    acc[0][2] = wmma_f16(a0, b2, acc[0][2]);
    acc[1][2] = wmma_f16(a1, b2, acc[1][2]);
    acc[0][3] = wmma_f16(a0, b3, acc[0][3]);
    acc[1][3] = wmma_f16(a1, b3, acc[1][3]);
  }
#pragma unroll
  for (int j = 0; j < 4; ++j) {
    int n = n0 + j * 16 + lo;
    float bn = bias ? bias[n] : 0.f;
#pragma unroll
    for (int i = 0; i < 2; ++i) {
      U8F c; c.v = acc[i][j];
#pragma unroll
      for (int r = 0; r < 8; ++r) {
        int m = m0 + i * 16 + r + 8 * hi;
        size_t o = (size_t)m * N + n;
        float v = c.e[r];
        if (epi == 0) {
          outH[o] = (_Float16)v;
        } else if (epi == 1) {
          outF[o] = v + bn + res[o];
        } else {
          float xx = v + bn;
          outH[o] = (_Float16)(0.5f * xx * (1.0f + erff(xx * 0.70710678118654752f)));
        }
      }
    }
  }
  (void)M;
}

// ---------------------------------------------------------------------------
// K2: Gram-tile pass 1 -> 5-NN density + per-batch dist max. 1 wave / 16 rows.
// ---------------------------------------------------------------------------
__global__ void knn_density_kernel(const _Float16* __restrict__ xnH, const float* __restrict__ sq,
                                   const float* __restrict__ noise, float* __restrict__ density,
                                   unsigned int* __restrict__ distmaxU) {
  __shared__ float tile[256];
  int lane = threadIdx.x & 31, lo = lane & 15, hi = lane >> 4;
  int b = blockIdx.y, i0 = blockIdx.x * 16;
  size_t bN = (size_t)b * STM_N;
  const _Float16* arow = xnH + (bN + i0 + lo) * STM_C;
  v16h afr[12];
  for (int kk = 0; kk < 12; ++kk) afr[kk] = load_a16(arow, kk * 32, hi);
  float sm[8];
  for (int r = 0; r < 8; ++r) sm[r] = sq[bN + i0 + r + 8 * hi];
  float tk[5] = {POS_INF, POS_INF, POS_INF, POS_INF, POS_INF};
  float rmax = 0.f;
  for (int jt = 0; jt < STM_N / 16; ++jt) {
    int n = jt * 16 + lo;
    const _Float16* bcol = xnH + (bN + n) * STM_C;
    __builtin_prefetch(bcol + 16 * STM_C, 0, 1);
    v8f acc = {};
    for (int kk = 0; kk < 12; ++kk) acc = wmma_f16(afr[kk], load_b16(bcol, kk * 32, hi), acc);
    float sqn = sq[bN + n];
    U8F c; c.v = acc;
    for (int r = 0; r < 8; ++r) {
      float d2 = sm[r] + sqn - 2.0f * c.e[r];
      tile[(r + 8 * hi) * 16 + lo] = sqrtf(fmaxf(d2, 0.f)) * INV_SQRT_DIM;
    }
    __syncthreads();
    if (lane < 16) {
      for (int q = 0; q < 16; ++q) {
        float v = tile[lane * 16 + q];
        rmax = fmaxf(rmax, v);
        if (v < tk[4]) {
          tk[4] = v;
          for (int s = 4; s > 0 && tk[s] < tk[s - 1]; --s) {
            float tmp = tk[s - 1]; tk[s - 1] = tk[s]; tk[s] = tmp;
          }
        }
      }
    }
    __syncthreads();
  }
  if (lane < 16) {
    int i = i0 + lane;
    float msq = (tk[0]*tk[0] + tk[1]*tk[1] + tk[2]*tk[2] + tk[3]*tk[3] + tk[4]*tk[4]) * 0.2f;
    density[bN + i] = expf(-msq) + noise[bN + i] * 1e-6f;
    atomicMax(&distmaxU[b], __float_as_uint(rmax));  // dist >= 0 -> uint order == float order
  }
}

// ---------------------------------------------------------------------------
// K3: Gram-tile pass 2 -> masked dmin, score = dmin * density.
// ---------------------------------------------------------------------------
__global__ void dmin_score_kernel(const _Float16* __restrict__ xnH, const float* __restrict__ sq,
                                  const float* __restrict__ density,
                                  const unsigned int* __restrict__ distmaxU,
                                  float* __restrict__ score) {
  int lane = threadIdx.x & 31, lo = lane & 15, hi = lane >> 4;
  int b = blockIdx.y, i0 = blockIdx.x * 16;
  size_t bN = (size_t)b * STM_N;
  const _Float16* arow = xnH + (bN + i0 + lo) * STM_C;
  v16h afr[12];
  for (int kk = 0; kk < 12; ++kk) afr[kk] = load_a16(arow, kk * 32, hi);
  float sm[8], dr[8], rmin[8];
  for (int r = 0; r < 8; ++r) {
    sm[r] = sq[bN + i0 + r + 8 * hi];
    dr[r] = density[bN + i0 + r + 8 * hi];
    rmin[r] = POS_INF;
  }
  float dmax = __uint_as_float(distmaxU[b]);
  for (int jt = 0; jt < STM_N / 16; ++jt) {
    int n = jt * 16 + lo;
    const _Float16* bcol = xnH + (bN + n) * STM_C;
    __builtin_prefetch(bcol + 16 * STM_C, 0, 1);
    v8f acc = {};
    for (int kk = 0; kk < 12; ++kk) acc = wmma_f16(afr[kk], load_b16(bcol, kk * 32, hi), acc);
    float sqn = sq[bN + n];
    float dn = density[bN + n];
    U8F c; c.v = acc;
    for (int r = 0; r < 8; ++r) {
      float d2 = sm[r] + sqn - 2.0f * c.e[r];
      float d = sqrtf(fmaxf(d2, 0.f)) * INV_SQRT_DIM;
      float val = (dn > dr[r]) ? d : dmax;
      rmin[r] = fminf(rmin[r], val);
    }
  }
  for (int r = 0; r < 8; ++r)
    for (int m = 1; m < 16; m <<= 1) rmin[r] = fminf(rmin[r], __shfl_xor(rmin[r], m, 32));
  if (lo == 0) {
    for (int r = 0; r < 8; ++r) {
      int i = i0 + r + 8 * hi;
      score[bN + i] = rmin[r] * density[bN + i];
    }
  }
}

// ---------------------------------------------------------------------------
// K4: per-batch full bitonic sort of 2048 scores (desc, idx-asc ties) -> top 512
// ---------------------------------------------------------------------------
__global__ void topk_kernel(const float* __restrict__ score, int* __restrict__ index_down) {
  __shared__ float key[STM_N];
  __shared__ int   idx[STM_N];
  int b = blockIdx.x, tid = threadIdx.x;
  for (int t = tid; t < STM_N; t += 1024) { key[t] = score[(size_t)b * STM_N + t]; idx[t] = t; }
  __syncthreads();
  for (int k = 2; k <= STM_N; k <<= 1) {
    for (int j = k >> 1; j > 0; j >>= 1) {
      for (int t = tid; t < STM_N; t += 1024) {
        int ixj = t ^ j;
        if (ixj > t) {
          bool up = ((t & k) == 0);
          float ka = key[t], kb = key[ixj];
          int   ia = idx[t], ib = idx[ixj];
          bool bBetter = (kb > ka) || (kb == ka && ib < ia);
          if (bBetter == up) { key[t] = kb; key[ixj] = ka; idx[t] = ib; idx[ixj] = ia; }
        }
      }
      __syncthreads();
    }
  }
  for (int t = tid; t < STM_OT; t += 1024) index_down[(size_t)b * STM_OT + t] = idx[t];
}

// ---------------------------------------------------------------------------
// K5: cluster assignment: idx_cluster[n] = argmin_i dist(index_down[i], n)
// (first-index tie semantics). 1 wave per 16-column stripe.
// ---------------------------------------------------------------------------
__global__ void assign_kernel(const _Float16* __restrict__ xnH, const float* __restrict__ sq,
                              const int* __restrict__ index_down, int* __restrict__ idx_cluster) {
  int lane = threadIdx.x & 31, lo = lane & 15, hi = lane >> 4;
  int b = blockIdx.y, n0 = blockIdx.x * 16;
  size_t bN = (size_t)b * STM_N;
  const _Float16* bcol = xnH + (bN + n0 + lo) * STM_C;
  v16h bfr[12];
  for (int kk = 0; kk < 12; ++kk) bfr[kk] = load_b16(bcol, kk * 32, hi);
  float sqn = sq[bN + n0 + lo];
  float bestd = POS_INF; int besti = 0;
  for (int i0 = 0; i0 < STM_OT; i0 += 16) {
    int rown = index_down[(size_t)b * STM_OT + i0 + lo];
    const _Float16* arow = xnH + (bN + rown) * STM_C;
    v8f acc = {};
    for (int kk = 0; kk < 12; ++kk) acc = wmma_f16(load_a16(arow, kk * 32, hi), bfr[kk], acc);
    U8F c; c.v = acc;
    for (int r = 0; r < 8; ++r) {
      int ii = i0 + r + 8 * hi;
      float smr = sq[bN + index_down[(size_t)b * STM_OT + ii]];
      float d2 = smr + sqn - 2.0f * c.e[r];
      float d = sqrtf(fmaxf(d2, 0.f)) * INV_SQRT_DIM;
      if (d < bestd) { bestd = d; besti = ii; }   // rows ascend -> first-min kept
    }
  }
  float od = __shfl_xor(bestd, 16, 32);
  int   oi = __shfl_xor(besti, 16, 32);
  if (od < bestd || (od == bestd && oi < besti)) { bestd = od; besti = oi; }
  if (lane < 16) idx_cluster[bN + n0 + lane] = besti;
}

__global__ void scatter_ids_kernel(const int* __restrict__ index_down, int* __restrict__ idx_cluster) {
  int b = blockIdx.x, t = threadIdx.x;
  idx_cluster[(size_t)b * STM_N + index_down[(size_t)b * STM_OT + t]] = t;
}

// ---------------------------------------------------------------------------
// K6: token merger (segment sums with exp(score) weights)
// ---------------------------------------------------------------------------
__global__ void segsum_w_kernel(const float* __restrict__ tscore, const int* __restrict__ idx_cluster,
                                float* __restrict__ all_w) {
  long t = (long)blockIdx.x * 256 + threadIdx.x;
  if (t >= (long)STM_B * STM_N) return;
  int b = (int)(t >> 11);
  atomicAdd(&all_w[(size_t)b * STM_OT + idx_cluster[t]], expf(tscore[t]));
}
__global__ void merge_feat_kernel(const _Float16* __restrict__ xnH, const float* __restrict__ tscore,
                                  const int* __restrict__ idx_cluster, const float* __restrict__ all_w,
                                  float* __restrict__ mergedF) {
  long row = blockIdx.x;
  int c = threadIdx.x;
  int b = (int)(row >> 11);
  int seg = b * STM_OT + idx_cluster[row];
  float w = expf(tscore[row]);
  float nw = w / (all_w[seg] + 1e-6f);
  atomicAdd(&mergedF[(size_t)seg * STM_C + c], (float)xnH[(size_t)row * STM_C + c] * nw);
}

// ---------------------------------------------------------------------------
// K9: flash attention. 1 wave per (b, h, 16 q-rows); online softmax over N.
// V tiles (32 keys x 64 dh) are staged to LDS with gfx1250 async global->LDS
// transfers (ASYNCcnt), overlapped with the QK^T WMMAs and softmax update.
// logits = (q . k) * dh^-0.5 + token_score[key]
// ---------------------------------------------------------------------------
__global__ void attn_kernel(const _Float16* __restrict__ qh, const _Float16* __restrict__ kvh,
                            const float* __restrict__ tscore, _Float16* __restrict__ attnH) {
  __shared__ _Float16 plds[16 * 32];
  __shared__ _Float16 vlds[32 * 64];
  int lane = threadIdx.x & 31, lo = lane & 15, hi = lane >> 4;
  int m0 = blockIdx.x * 16, h = blockIdx.y, b = blockIdx.z;
  size_t bN = (size_t)b * STM_N;
  const _Float16* qrow = qh + ((size_t)b * STM_OT + m0 + lo) * STM_C + h * STM_DH;
  v16h qa0 = load_a16(qrow, 0, hi);
  v16h qa1 = load_a16(qrow, 32, hi);
  v8f acc[4] = {};
  float mrun[8], lrun[8];
  for (int r = 0; r < 8; ++r) { mrun[r] = -POS_INF; lrun[r] = 0.f; }
  const float scale = 0.125f;  // 64^-0.5
  // LDS byte address = low 32 bits of the flat shared pointer (ISA 10.2).
  unsigned vbase = (unsigned)(unsigned long long)&vlds[0] + (unsigned)lane * 128u;
  for (int nb = 0; nb < STM_N; nb += 32) {
    __syncthreads();  // prior iteration's LDS reads complete before overwrite
    {   // async-stage this iteration's V tile: lane -> key row nb+lane, 128B
      const char* vrow = (const char*)(kvh + (bN + nb + lane) * (2 * STM_C) + STM_C + h * STM_DH);
      __builtin_prefetch(vrow + 32 * 2 * STM_C * 2, 0, 1);
#pragma unroll
      for (int q = 0; q < 8; ++q) {
        unsigned dst = vbase + q * 16u;
        const char* src = vrow + q * 16;
        asm volatile("global_load_async_to_lds_b128 %0, %1, off"
                     :: "v"(dst), "v"(src) : "memory");
      }
    }
    U8F lt[2];
    for (int t = 0; t < 2; ++t) {
      int n = nb + t * 16 + lo;
      const _Float16* kc = kvh + (bN + n) * (2 * STM_C) + h * STM_DH;
      __builtin_prefetch(kc + 32 * 2 * STM_C, 0, 1);
      v8f a = {};
      a = wmma_f16(qa0, load_b16(kc, 0, hi), a);
      a = wmma_f16(qa1, load_b16(kc, 32, hi), a);
      float ts = tscore[bN + n];
      lt[t].v = a;
      for (int r = 0; r < 8; ++r) lt[t].e[r] = lt[t].e[r] * scale + ts;
    }
    float corr[8];
    for (int r = 0; r < 8; ++r) {
      float v = fmaxf(lt[0].e[r], lt[1].e[r]);
      for (int m = 1; m < 16; m <<= 1) v = fmaxf(v, __shfl_xor(v, m, 32));
      float nm = fmaxf(mrun[r], v);
      corr[r] = expf(mrun[r] - nm);
      mrun[r] = nm;
    }
    for (int r = 0; r < 8; ++r) {
      float p0 = expf(lt[0].e[r] - mrun[r]);
      float p1 = expf(lt[1].e[r] - mrun[r]);
      lt[0].e[r] = p0; lt[1].e[r] = p1;
      float s = p0 + p1;
      for (int m = 1; m < 16; m <<= 1) s += __shfl_xor(s, m, 32);
      lrun[r] = lrun[r] * corr[r] + s;
    }
    for (int f = 0; f < 4; ++f) {
      U8F a; a.v = acc[f];
      for (int r = 0; r < 8; ++r) a.e[r] *= corr[r];
      acc[f] = a.v;
    }
    for (int r = 0; r < 8; ++r) {
      int rr = r + 8 * hi;
      plds[rr * 32 + lo]      = (_Float16)lt[0].e[r];
      plds[rr * 32 + 16 + lo] = (_Float16)lt[1].e[r];
    }
    asm volatile("s_wait_asynccnt 0" ::: "memory");  // V tile resident in LDS
    __syncthreads();
    U16H pf;  // C-layout -> A-layout transpose through LDS
    for (int e = 0; e < 16; ++e) {
      int k = hi * 16 + e;
      pf.e[e] = plds[lo * 32 + k];
    }
    for (int f = 0; f < 4; ++f) {
      U16H vf;  // B[k=n_local][col=d] from the LDS-staged V tile
      for (int e = 0; e < 16; ++e) {
        vf.e[e] = vlds[(hi * 16 + e) * 64 + f * 16 + lo];
      }
      acc[f] = wmma_f16(pf.v, vf.v, acc[f]);
    }
  }
  float invl[8];
  for (int r = 0; r < 8; ++r) invl[r] = 1.0f / lrun[r];
  for (int f = 0; f < 4; ++f) {
    U8F a; a.v = acc[f];
    for (int r = 0; r < 8; ++r) {
      size_t o = ((size_t)b * STM_OT + m0 + r + 8 * hi) * STM_C + h * STM_DH + f * 16 + lo;
      attnH[o] = (_Float16)(a.e[r] * invl[r]);
    }
  }
}

// ---------------------------------------------------------------------------
// Host: launch pipeline on `stream` (graph-capture safe: kernels only).
// ---------------------------------------------------------------------------
extern "C" void kernel_launch(void* const* d_in, const int* in_sizes, int n_in,
                              void* d_out, int out_size, void* d_ws, size_t ws_size,
                              hipStream_t stream) {
  const float* x       = (const float*)d_in[0];
  const float* noise   = (const float*)d_in[1];
  const float* n1g     = (const float*)d_in[2];
  const float* n1b     = (const float*)d_in[3];
  const float* score_w = (const float*)d_in[4];
  const float* score_b = (const float*)d_in[5];
  const float* q_w     = (const float*)d_in[6];
  const float* kv_w    = (const float*)d_in[7];
  const float* proj_w  = (const float*)d_in[8];
  const float* proj_b  = (const float*)d_in[9];
  const float* n2g     = (const float*)d_in[10];
  const float* n2b     = (const float*)d_in[11];
  const float* fc1_w   = (const float*)d_in[12];
  const float* fc1_b   = (const float*)d_in[13];
  const float* fc2_w   = (const float*)d_in[14];
  const float* fc2_b   = (const float*)d_in[15];
  float* out = (float*)d_out;
  (void)in_sizes; (void)n_in; (void)out_size; (void)ws_size;

  const int B = STM_B, N = STM_N, C = STM_C, OT = STM_OT, FC = STM_FC;
  char* base = (char*)d_ws;
  size_t off = 0;
  auto alloc = [&](size_t bytes) -> char* {
    char* p = base + off;
    off += (bytes + 255) & ~(size_t)255;
    return p;
  };
  _Float16* xnH     = (_Float16*)alloc((size_t)B * N * C * 2);
  float*    tscore  = (float*)alloc((size_t)B * N * 4);
  float*    sq      = (float*)alloc((size_t)B * N * 4);
  float*    density = (float*)alloc((size_t)B * N * 4);
  unsigned* distmax = (unsigned*)alloc((size_t)B * 4);
  float*    score   = (float*)alloc((size_t)B * N * 4);
  int*      idxdown = (int*)alloc((size_t)B * OT * 4);
  int*      idxclus = (int*)alloc((size_t)B * N * 4);
  float*    all_w   = (float*)alloc((size_t)B * OT * 4);
  float*    mergedF = (float*)alloc((size_t)B * OT * C * 4);
  _Float16* mergedH = (_Float16*)alloc((size_t)B * OT * C * 2);
  _Float16* qH      = (_Float16*)alloc((size_t)B * OT * C * 2);
  _Float16* kvH     = (_Float16*)alloc((size_t)B * N * 2 * C * 2);
  _Float16* attnH   = (_Float16*)alloc((size_t)B * OT * C * 2);
  float*    featF   = (float*)alloc((size_t)B * OT * C * 4);
  _Float16* ln2H    = (_Float16*)alloc((size_t)B * OT * C * 2);
  _Float16* hmH     = (_Float16*)alloc((size_t)B * OT * FC * 2);
  _Float16* qwT     = (_Float16*)alloc((size_t)C * C * 2);
  _Float16* kvwT    = (_Float16*)alloc((size_t)C * 2 * C * 2);
  _Float16* projwT  = (_Float16*)alloc((size_t)C * C * 2);
  _Float16* fc1T    = (_Float16*)alloc((size_t)C * FC * 2);
  _Float16* fc2T    = (_Float16*)alloc((size_t)FC * C * 2);

  auto g1 = [](long n) { return dim3((unsigned)((n + 255) / 256)); };

  // Weight fp16 transposes ([K,N] -> [N,K] for contiguous B-fragments)
  wtr_kernel<<<g1((long)C * C), 256, 0, stream>>>(q_w, qwT, C, C);
  wtr_kernel<<<g1((long)C * 2 * C), 256, 0, stream>>>(kv_w, kvwT, C, 2 * C);
  wtr_kernel<<<g1((long)C * C), 256, 0, stream>>>(proj_w, projwT, C, C);
  wtr_kernel<<<g1((long)C * FC), 256, 0, stream>>>(fc1_w, fc1T, C, FC);
  wtr_kernel<<<g1((long)FC * C), 256, 0, stream>>>(fc2_w, fc2T, FC, C);

  // LN1 + score + sq
  ln_score_kernel<<<dim3(B * N / 8), 256, 0, stream>>>(x, n1g, n1b, score_w, score_b,
                                                       xnH, tscore, sq);
  // Zero accumulators
  zero_kernel<<<g1(B), 256, 0, stream>>>((float*)distmax, B);
  zero_kernel<<<g1((long)B * OT), 256, 0, stream>>>(all_w, (long)B * OT);
  zero_kernel<<<g1((long)B * OT * C), 256, 0, stream>>>(mergedF, (long)B * OT * C);

  // Clustering
  knn_density_kernel<<<dim3(N / 16, B), 32, 0, stream>>>(xnH, sq, noise, density, distmax);
  dmin_score_kernel<<<dim3(N / 16, B), 32, 0, stream>>>(xnH, sq, density, distmax, score);
  topk_kernel<<<dim3(B), 1024, 0, stream>>>(score, idxdown);
  assign_kernel<<<dim3(N / 16, B), 32, 0, stream>>>(xnH, sq, idxdown, idxclus);
  scatter_ids_kernel<<<dim3(B), OT, 0, stream>>>(idxdown, idxclus);

  // Merger
  segsum_w_kernel<<<g1((long)B * N), 256, 0, stream>>>(tscore, idxclus, all_w);
  merge_feat_kernel<<<dim3(B * N), C, 0, stream>>>(xnH, tscore, idxclus, all_w, mergedF);
  f2h_kernel<<<g1((long)B * OT * C), 256, 0, stream>>>(mergedF, mergedH, (long)B * OT * C);

  // Projections (blocked GEMM: grid = (N/64, M/32))
  gemm_wmma_kernel<<<dim3(C / 64, B * OT / 32), 32, 0, stream>>>(
      mergedH, qwT, B * OT, C, C, nullptr, qH, nullptr, nullptr, 0);
  gemm_wmma_kernel<<<dim3(2 * C / 64, B * N / 32), 32, 0, stream>>>(
      xnH, kvwT, B * N, 2 * C, C, nullptr, kvH, nullptr, nullptr, 0);

  // Attention
  attn_kernel<<<dim3(OT / 16, STM_H, B), 32, 0, stream>>>(qH, kvH, tscore, attnH);

  // proj + residual -> feat
  gemm_wmma_kernel<<<dim3(C / 64, B * OT / 32), 32, 0, stream>>>(
      attnH, projwT, B * OT, C, C, featF, nullptr, proj_b, mergedF, 1);

  // MLP
  ln2_kernel<<<dim3(B * OT / 8), 256, 0, stream>>>(featF, n2g, n2b, ln2H);
  gemm_wmma_kernel<<<dim3(FC / 64, B * OT / 32), 32, 0, stream>>>(
      ln2H, fc1T, B * OT, FC, C, nullptr, hmH, fc1_b, nullptr, 2);
  gemm_wmma_kernel<<<dim3(C / 64, B * OT / 32), 32, 0, stream>>>(
      hmH, fc2T, B * OT, C, FC, out, nullptr, fc2_b, featF, 1);
}
